// QNetwork_10651518894533
// MI455X (gfx1250) — matrix operand order repaired and checked by reading, not compiled, not run on hardware
//
#include <hip/hip_runtime.h>
#include <hip/hip_bf16.h>

// ---------------------------------------------------------------------------
// MI455X fused GraphNetwork QNetwork.
//  - All GEMMs via v_wmma_f32_16x16x32_bf16 (wave32, 16x16x32 bf16 -> f32).
//  - Edge pipeline fully fused: e1 (K=32 WMMA) lives only in LDS, feeds
//    e2 (K=256 WMMA); both blocks' segment sums scattered with f32 atomics
//    whose targets fit in the 192MB L2.
//  - Weights pre-converted to bf16, stacked (multi-input updates become one
//    long-K GEMM), and pre-swizzled into WMMA B-fragment order so each
//    fragment is one aligned 32B/lane load (2x global_load_b128, coalesced).
//  - edge_graph/node_graph are sorted: uniform-tile fast path does a lane-
//    local 8-row reduction and a single atomic per column (8x fewer atomics).
//  - 32-bit unsigned offsets for atomics -> saddr+voffset addressing.
// ---------------------------------------------------------------------------

#define N_NODES  50000
#define N_EDGES  500000
#define N_GRAPHS 512

typedef __attribute__((ext_vector_type(16))) __bf16 v16bf;
typedef __attribute__((ext_vector_type(8)))  float  v8f;

union BF16x16 { v16bf v; unsigned short u[16]; };
union F32x8   { v8f   v; float          f[8];  };

__device__ __forceinline__ unsigned short f2bf(float f) {
    union { float f; unsigned u; } x; x.f = f;
    unsigned u = x.u;
    unsigned r = (u + 0x7FFFu + ((u >> 16) & 1u)) >> 16;  // RNE bf16
    return (unsigned short)r;
}

__device__ __forceinline__ int imax1(int a) { return a > 1 ? a : 1; }

// A-fragment (16x32 bf16) from row-major LDS tile, leading dim ldk, k offset k0.
// ISA 7.12.2: lanes 0-15 M=0..15; VGPR j: j<4 -> K=8g+2j, j>=4 -> K=8g+16+2(j-4)
__device__ __forceinline__ v16bf load_afrag_lds(const unsigned short* As, int ldk,
                                                int k0, int lane) {
    int r = lane & 15, g = lane >> 4;
    BF16x16 a;
#pragma unroll
    for (int j = 0; j < 8; ++j) {
        int k = k0 + 8 * g + ((j < 4) ? 2 * j : 16 + 2 * (j - 4));
        a.u[2 * j]     = As[r * ldk + k];
        a.u[2 * j + 1] = As[r * ldk + k + 1];
    }
    return a.v;
}

// B-fragment from pre-swizzled weights: layout [n_tile][k_step][lane][16] bf16.
// One aligned 32-byte load per lane (lowers to 2x global_load_b128).
__device__ __forceinline__ v16bf load_bfrag_sw(const unsigned short* __restrict__ Wf,
                                               int nks, int nt, int ks, int lane) {
    return *(const v16bf*)(Wf + ((((size_t)nt * nks + ks) * 32 + lane) << 4));
}

__device__ __forceinline__ void wmma_bf16(F32x8& c, v16bf a, v16bf b) {
    c.v = __builtin_amdgcn_wmma_f32_16x16x32_bf16(
        false, a, false, b, (short)0, c.v, false, false);
}

// ---------------------------------------------------------------------------
// setup: f32 -> bf16 conversion, then swizzle into B-fragment order
// ---------------------------------------------------------------------------
__global__ void cvt_kernel(const float* __restrict__ src,
                           unsigned short* __restrict__ dst, int n) {
    int i = blockIdx.x * blockDim.x + threadIdx.x;
    if (i < n) dst[i] = f2bf(src[i]);
}

// W: plain row-major bf16 [K][N] -> Wf: [n_tile][k_step][lane][16]
__global__ void swz_kernel(const unsigned short* __restrict__ W,
                           unsigned short* __restrict__ Wf, int K, int N) {
    int i = blockIdx.x * blockDim.x + threadIdx.x;
    if (i >= K * N) return;
    int j    = i & 15;
    int lane = (i >> 4) & 31;
    int rest = i >> 9;                 // nt*nks + ks
    int nks  = K >> 5;
    int ks = rest % nks, nt = rest / nks;
    int n = nt * 16 + (lane & 15);
    int g = lane >> 4;
    int k = ks * 32 + 8 * g + ((j < 8) ? j : 16 + (j - 8));
    Wf[i] = W[(size_t)k * N + n];
}

// ---------------------------------------------------------------------------
// degree / graph-size counts (int atomics; tiny L2-resident targets)
// ---------------------------------------------------------------------------
__global__ void count_kernel(const int* __restrict__ senders,
                             const int* __restrict__ receivers,
                             const int* __restrict__ edge_graph,
                             const int* __restrict__ node_graph,
                             int* indeg, int* outdeg, int* gecnt, int* gncnt) {
    int i = blockIdx.x * blockDim.x + threadIdx.x;
    if (i < N_EDGES) {
        atomicAdd(&outdeg[(unsigned)senders[i]], 1);
        atomicAdd(&indeg[(unsigned)receivers[i]], 1);
        atomicAdd(&gecnt[(unsigned)edge_graph[i]], 1);
    }
    if (i < N_NODES) atomicAdd(&gncnt[(unsigned)node_graph[i]], 1);
}

// ---------------------------------------------------------------------------
// Fused edge pipeline per 16-edge tile:
//   e1 = relu(e @ We1 + be1)   (WMMA K=32)  -> LDS only
//   scatter e1 -> inc1/out1/eg1 sums; e2 = relu(e1 @ We2 + be2) (K=256)
//   scatter e2 -> inc2/out2/eg2 sums. e1/e2 never touch HBM.
// ---------------------------------------------------------------------------
__global__ __launch_bounds__(256) void edge_kernel(
    const float* __restrict__ e,
    const int* __restrict__ senders, const int* __restrict__ receivers,
    const int* __restrict__ edge_graph,
    const unsigned short* __restrict__ We1f, const float* __restrict__ be1,
    const unsigned short* __restrict__ We2f, const float* __restrict__ be2,
    float* __restrict__ inc1, float* __restrict__ out1, float* __restrict__ eg1,
    float* __restrict__ inc2, float* __restrict__ out2, float* __restrict__ eg2) {
    __shared__ unsigned short es[16 * 32];
    __shared__ unsigned short e1s[16 * 256];
    __shared__ int sidx[16], ridx[16], gidx[16];
    __shared__ int guni;  // graph id if tile-uniform, else -1

    const int t = threadIdx.x, lane = t & 31, wave = t >> 5;
    const int ebase = blockIdx.x * 16;

    for (int i = t; i < 16 * 32; i += 256) {
        int r = i >> 5, c = i & 31;
        es[i] = f2bf(e[(size_t)(ebase + r) * 32 + c]);
    }
    if (t < 16) {
        sidx[t] = senders[ebase + t];
        ridx[t] = receivers[ebase + t];
        gidx[t] = edge_graph[ebase + t];
    }
    __builtin_prefetch(We2f, 0, 3);  // global_prefetch_b8: warm We2 fragments
    __syncthreads();
    if (t == 0) {
        // edge_graph is sorted: uniform iff first == last
        guni = (gidx[0] == gidx[15]) ? gidx[0] : -1;
    }
    __syncthreads();

    const int g = lane >> 4, col = lane & 15;
    const int gu = guni;

    // stage B: e1; wave w owns n-tiles 2w,2w+1 of 16 (256 cols)
    v16bf afr = load_afrag_lds(es, 32, 0, lane);
#pragma unroll
    for (int q = 0; q < 2; ++q) {
        int n0 = (wave * 2 + q) * 16;
        F32x8 c;
#pragma unroll
        for (int i = 0; i < 8; ++i) c.f[i] = 0.f;
        wmma_bf16(c, afr, load_bfrag_sw(We1f, 1, wave * 2 + q, 0, lane));
        unsigned n = (unsigned)(n0 + col);
        float bias = be1[n];
        float vv[8];
#pragma unroll
        for (int i = 0; i < 8; ++i) {
            float v = c.f[i] + bias;
            vv[i] = v > 0.f ? v : 0.f;
            e1s[(i + 8 * g) * 256 + n] = f2bf(vv[i]);
        }
#pragma unroll
        for (int i = 0; i < 8; ++i) {
            int row = i + 8 * g;
            atomicAdd(&inc1[(unsigned)ridx[row] * 256u + n], vv[i]);
            atomicAdd(&out1[(unsigned)sidx[row] * 256u + n], vv[i]);
        }
        if (gu >= 0) {  // sorted graphs: one atomic per column
            float s = 0.f;
#pragma unroll
            for (int i = 0; i < 8; ++i) s += vv[i];
            atomicAdd(&eg1[(unsigned)gu * 256u + n], s);
        } else {
#pragma unroll
            for (int i = 0; i < 8; ++i)
                atomicAdd(&eg1[(unsigned)gidx[i + 8 * g] * 256u + n], vv[i]);
        }
    }
    __syncthreads();

    // stage C: e2 = relu(e1s @ We2 + be2); wave w owns n-tile w of 8 (128 cols)
    {
        int n0 = wave * 16;
        F32x8 c;
#pragma unroll
        for (int i = 0; i < 8; ++i) c.f[i] = 0.f;
#pragma unroll
        for (int ks = 0; ks < 8; ++ks)
            wmma_bf16(c, load_afrag_lds(e1s, 256, ks * 32, lane),
                      load_bfrag_sw(We2f, 8, wave, ks, lane));
        unsigned n = (unsigned)(n0 + col);
        float bias = be2[n];
        float vv[8];
#pragma unroll
        for (int i = 0; i < 8; ++i) {
            float v = c.f[i] + bias;
            vv[i] = v > 0.f ? v : 0.f;
        }
#pragma unroll
        for (int i = 0; i < 8; ++i) {
            int row = i + 8 * g;
            atomicAdd(&inc2[(unsigned)ridx[row] * 128u + n], vv[i]);
            atomicAdd(&out2[(unsigned)sidx[row] * 128u + n], vv[i]);
        }
        if (gu >= 0) {
            float s = 0.f;
#pragma unroll
            for (int i = 0; i < 8; ++i) s += vv[i];
            atomicAdd(&eg2[(unsigned)gu * 128u + n], s);
        } else {
#pragma unroll
            for (int i = 0; i < 8; ++i)
                atomicAdd(&eg2[(unsigned)gidx[i + 8 * g] * 128u + n], vv[i]);
        }
    }
}

// ---------------------------------------------------------------------------
// Node block 1: n1 = relu([x | mean(inc1) | mean(out1)] @ [Wn1;Win1;Wout1]+bn1)
// K=576 GEMM. Stores n1 bf16, scatters graph sum (uniform fast path).
// ---------------------------------------------------------------------------
__global__ __launch_bounds__(256) void node1_kernel(
    const float* __restrict__ x, const int* __restrict__ node_graph,
    const float* __restrict__ inc1, const float* __restrict__ out1,
    const int* __restrict__ indeg, const int* __restrict__ outdeg,
    const unsigned short* __restrict__ Wn1f, const float* __restrict__ bn1,
    unsigned short* __restrict__ n1, float* __restrict__ ng1) {
    __shared__ unsigned short As[16 * 576];
    __shared__ int gidx[16];
    __shared__ int guni;
    const int t = threadIdx.x, lane = t & 31, wave = t >> 5;
    const int nbase = blockIdx.x * 16;

    if (t < 16) gidx[t] = node_graph[nbase + t];
    for (int i = t; i < 16 * 64; i += 256) {
        int r = i >> 6, c = i & 63;
        As[r * 576 + c] = f2bf(x[(size_t)(nbase + r) * 64 + c]);
    }
    for (int i = t; i < 16 * 256; i += 256) {
        int r = i >> 8, c = i & 255;
        unsigned node = (unsigned)(nbase + r);
        As[r * 576 + 64 + c]  = f2bf(inc1[node * 256u + c] / (float)imax1(indeg[node]));
        As[r * 576 + 320 + c] = f2bf(out1[node * 256u + c] / (float)imax1(outdeg[node]));
    }
    __syncthreads();
    if (t == 0) guni = (gidx[0] == gidx[15]) ? gidx[0] : -1;
    __syncthreads();

    const int g = lane >> 4, col = lane & 15;
    const int gu = guni;
#pragma unroll
    for (int q = 0; q < 2; ++q) {
        int nt = wave * 2 + q;
        F32x8 c;
#pragma unroll
        for (int i = 0; i < 8; ++i) c.f[i] = 0.f;
        for (int ks = 0; ks < 18; ++ks)
            wmma_bf16(c, load_afrag_lds(As, 576, ks * 32, lane),
                      load_bfrag_sw(Wn1f, 18, nt, ks, lane));
        unsigned n = (unsigned)(nt * 16 + col);
        float bias = bn1[n];
        float vv[8];
#pragma unroll
        for (int i = 0; i < 8; ++i) {
            float v = c.f[i] + bias;
            vv[i] = v > 0.f ? v : 0.f;
            n1[(unsigned)(nbase + i + 8 * g) * 256u + n] = f2bf(vv[i]);
        }
        if (gu >= 0) {
            float s = 0.f;
#pragma unroll
            for (int i = 0; i < 8; ++i) s += vv[i];
            atomicAdd(&ng1[(unsigned)gu * 256u + n], s);
        } else {
#pragma unroll
            for (int i = 0; i < 8; ++i)
                atomicAdd(&ng1[(unsigned)gidx[i + 8 * g] * 256u + n], vv[i]);
        }
    }
}

// ---------------------------------------------------------------------------
// Node block 2: n2 = relu([n1 | mean(inc2) | mean(out2)] @ [Wn2;Win2;Wout2]+bn2)
// K=512; n2 only feeds its graph-mean -> scatter, never stored.
// ---------------------------------------------------------------------------
__global__ __launch_bounds__(256) void node2_kernel(
    const unsigned short* __restrict__ n1, const int* __restrict__ node_graph,
    const float* __restrict__ inc2, const float* __restrict__ out2,
    const int* __restrict__ indeg, const int* __restrict__ outdeg,
    const unsigned short* __restrict__ Wn2f, const float* __restrict__ bn2,
    float* __restrict__ ng2) {
    __shared__ unsigned short As[16 * 512];
    __shared__ int gidx[16];
    __shared__ int guni;
    const int t = threadIdx.x, lane = t & 31, wave = t >> 5;
    const int nbase = blockIdx.x * 16;

    if (t < 16) gidx[t] = node_graph[nbase + t];
    for (int i = t; i < 16 * 256; i += 256) {
        int r = i >> 8, c = i & 255;
        As[r * 512 + c] = n1[(unsigned)(nbase + r) * 256u + c];
    }
    for (int i = t; i < 16 * 128; i += 256) {
        int r = i >> 7, c = i & 127;
        unsigned node = (unsigned)(nbase + r);
        As[r * 512 + 256 + c] = f2bf(inc2[node * 128u + c] / (float)imax1(indeg[node]));
        As[r * 512 + 384 + c] = f2bf(out2[node * 128u + c] / (float)imax1(outdeg[node]));
    }
    __syncthreads();
    if (t == 0) guni = (gidx[0] == gidx[15]) ? gidx[0] : -1;
    __syncthreads();

    const int g = lane >> 4, col = lane & 15;
    const int gu = guni;
    F32x8 c;
#pragma unroll
    for (int i = 0; i < 8; ++i) c.f[i] = 0.f;
    for (int ks = 0; ks < 16; ++ks)
        wmma_bf16(c, load_afrag_lds(As, 512, ks * 32, lane),
                  load_bfrag_sw(Wn2f, 16, wave, ks, lane));
    unsigned n = (unsigned)(wave * 16 + col);
    float bias = bn2[n];
    float vv[8];
#pragma unroll
    for (int i = 0; i < 8; ++i) {
        float v = c.f[i] + bias;
        vv[i] = v > 0.f ? v : 0.f;
    }
    if (gu >= 0) {
        float s = 0.f;
#pragma unroll
        for (int i = 0; i < 8; ++i) s += vv[i];
        atomicAdd(&ng2[(unsigned)gu * 128u + n], s);
    } else {
#pragma unroll
        for (int i = 0; i < 8; ++i)
            atomicAdd(&ng2[(unsigned)gidx[i + 8 * g] * 128u + n], vv[i]);
    }
}

// ---------------------------------------------------------------------------
// Global pipeline (512 graphs, 32 WGs x 16 graphs): u1 -> u2 -> readout -> MLP
// ---------------------------------------------------------------------------
__global__ __launch_bounds__(256) void global_kernel(
    const float* __restrict__ u, const float* __restrict__ aF,
    const float* __restrict__ ng1, const float* __restrict__ eg1,
    const float* __restrict__ ng2, const float* __restrict__ eg2,
    const int* __restrict__ gncnt, const int* __restrict__ gecnt,
    const unsigned short* __restrict__ Wg1f, const float* __restrict__ bg1,
    const unsigned short* __restrict__ Wg2f, const float* __restrict__ bg2,
    const unsigned short* __restrict__ Wgaf, const float* __restrict__ bga,
    const unsigned short* __restrict__ Wa1f, const float* __restrict__ ba1,
    const unsigned short* __restrict__ Wa2f, const float* __restrict__ ba2,
    const float* __restrict__ Wa3, const float* __restrict__ ba3,
    float* __restrict__ out) {
    __shared__ unsigned short As[16 * 576];   // reused: A1 (576), A2 (512)
    __shared__ unsigned short u1s[16 * 256];
    __shared__ unsigned short u2s[16 * 128];
    __shared__ unsigned short hs[16 * 160];
    __shared__ unsigned short h1s[16 * 256];
    __shared__ float h2f[16 * 256];
    __shared__ float red[16];

    const int t = threadIdx.x, lane = t & 31, wave = t >> 5;
    const int gbase = blockIdx.x * 16;
    const int g = lane >> 4, col = lane & 15;

    // ---- stage 1: u1 = relu([u | nbar1 | ebar1] @ [Wg1;Wgn1;Wge1] + bg1)
    for (int i = t; i < 16 * 64; i += 256) {
        int r = i >> 6, c = i & 63;
        As[r * 576 + c] = f2bf(u[(size_t)(gbase + r) * 64 + c]);
    }
    for (int i = t; i < 16 * 256; i += 256) {
        int r = i >> 8, c = i & 255;
        unsigned gr = (unsigned)(gbase + r);
        As[r * 576 + 64 + c]  = f2bf(ng1[gr * 256u + c] / (float)imax1(gncnt[gr]));
        As[r * 576 + 320 + c] = f2bf(eg1[gr * 256u + c] / (float)imax1(gecnt[gr]));
    }
    __syncthreads();
#pragma unroll
    for (int q = 0; q < 2; ++q) {
        int nt = wave * 2 + q;
        F32x8 c;
#pragma unroll
        for (int i = 0; i < 8; ++i) c.f[i] = 0.f;
        for (int ks = 0; ks < 18; ++ks)
            wmma_bf16(c, load_afrag_lds(As, 576, ks * 32, lane),
                      load_bfrag_sw(Wg1f, 18, nt, ks, lane));
        int n = nt * 16 + col;
        float bias = bg1[n];
#pragma unroll
        for (int i = 0; i < 8; ++i) {
            float v = c.f[i] + bias;
            u1s[(i + 8 * g) * 256 + n] = f2bf(v > 0.f ? v : 0.f);
        }
    }
    __syncthreads();

    // ---- stage 2: u2 = relu([u1 | nbar2 | ebar2] @ [Wg2;Wgn2;Wge2] + bg2)
    for (int i = t; i < 16 * 256; i += 256) {
        int r = i >> 8, c = i & 255;
        As[r * 512 + c] = u1s[r * 256 + c];
    }
    for (int i = t; i < 16 * 128; i += 256) {
        int r = i >> 7, c = i & 127;
        unsigned gr = (unsigned)(gbase + r);
        As[r * 512 + 256 + c] = f2bf(ng2[gr * 128u + c] / (float)imax1(gncnt[gr]));
        As[r * 512 + 384 + c] = f2bf(eg2[gr * 128u + c] / (float)imax1(gecnt[gr]));
    }
    __syncthreads();
    {
        F32x8 c;
#pragma unroll
        for (int i = 0; i < 8; ++i) c.f[i] = 0.f;
        for (int ks = 0; ks < 16; ++ks)
            wmma_bf16(c, load_afrag_lds(As, 512, ks * 32, lane),
                      load_bfrag_sw(Wg2f, 16, wave, ks, lane));
        int n = wave * 16 + col;
        float bias = bg2[n];
#pragma unroll
        for (int i = 0; i < 8; ++i) {
            float v = c.f[i] + bias;
            u2s[(i + 8 * g) * 128 + n] = f2bf(v > 0.f ? v : 0.f);
        }
    }
    // action + zero-pad columns of h
    for (int i = t; i < 16 * 32; i += 256) {
        int r = i >> 5, c = i & 31;
        hs[r * 160 + 128 + c] = (c < 8) ? f2bf(aF[(size_t)(gbase + r) * 8 + c])
                                        : (unsigned short)0;
    }
    __syncthreads();

    // ---- stage 3: state_value = u2 @ Wga + bga  (K=128), no relu
    {
        F32x8 c;
#pragma unroll
        for (int i = 0; i < 8; ++i) c.f[i] = 0.f;
        for (int ks = 0; ks < 4; ++ks)
            wmma_bf16(c, load_afrag_lds(u2s, 128, ks * 32, lane),
                      load_bfrag_sw(Wgaf, 4, wave, ks, lane));
        int n = wave * 16 + col;
        float bias = bga[n];
#pragma unroll
        for (int i = 0; i < 8; ++i)
            hs[(i + 8 * g) * 160 + n] = f2bf(c.f[i] + bias);
    }
    __syncthreads();

    // ---- stage 4: h1 = relu([sv | a | 0] @ Wa1p + ba1)  (K=160)
#pragma unroll
    for (int q = 0; q < 2; ++q) {
        int nt = wave * 2 + q;
        F32x8 c;
#pragma unroll
        for (int i = 0; i < 8; ++i) c.f[i] = 0.f;
        for (int ks = 0; ks < 5; ++ks)
            wmma_bf16(c, load_afrag_lds(hs, 160, ks * 32, lane),
                      load_bfrag_sw(Wa1f, 5, nt, ks, lane));
        int n = nt * 16 + col;
        float bias = ba1[n];
#pragma unroll
        for (int i = 0; i < 8; ++i) {
            float v = c.f[i] + bias;
            h1s[(i + 8 * g) * 256 + n] = f2bf(v > 0.f ? v : 0.f);
        }
    }
    __syncthreads();

    // ---- stage 5: h2 = relu(h1 @ Wa2 + ba2)  (K=256), keep f32 for final dot
#pragma unroll
    for (int q = 0; q < 2; ++q) {
        int nt = wave * 2 + q;
        F32x8 c;
#pragma unroll
        for (int i = 0; i < 8; ++i) c.f[i] = 0.f;
        for (int ks = 0; ks < 8; ++ks)
            wmma_bf16(c, load_afrag_lds(h1s, 256, ks * 32, lane),
                      load_bfrag_sw(Wa2f, 8, nt, ks, lane));
        int n = nt * 16 + col;
        float bias = ba2[n];
#pragma unroll
        for (int i = 0; i < 8; ++i) {
            float v = c.f[i] + bias;
            h2f[(i + 8 * g) * 256 + n] = v > 0.f ? v : 0.f;
        }
    }
    if (t < 16) red[t] = 0.f;
    __syncthreads();

    // ---- stage 6: out = h2 @ Wa3 + ba3  ([16,256]x[256,1], VALU reduction)
    {
        int row = t >> 4, c16 = t & 15;
        float p = 0.f;
#pragma unroll
        for (int j = 0; j < 16; ++j) {
            int k = c16 * 16 + j;
            p += h2f[row * 256 + k] * Wa3[k];
        }
        atomicAdd(&red[row], p);
    }
    __syncthreads();
    if (t < 16) out[gbase + t] = red[t] + ba3[0];
}

// ---------------------------------------------------------------------------
// host launcher
// ---------------------------------------------------------------------------
static inline size_t alignUp256(size_t x) { return (x + 255) & ~(size_t)255; }

extern "C" void kernel_launch(void* const* d_in, const int* in_sizes, int n_in,
                              void* d_out, int out_size, void* d_ws, size_t ws_size,
                              hipStream_t stream) {
    (void)in_sizes; (void)n_in; (void)out_size; (void)ws_size;
    const float* x          = (const float*)d_in[0];
    const float* e          = (const float*)d_in[1];
    const float* u          = (const float*)d_in[2];
    const float* aF         = (const float*)d_in[3];
    const int*   senders    = (const int*)d_in[4];
    const int*   receivers  = (const int*)d_in[5];
    const int*   node_graph = (const int*)d_in[6];
    const int*   edge_graph = (const int*)d_in[7];
    const float* We1  = (const float*)d_in[8];   const float* be1 = (const float*)d_in[9];
    const float* Wn1  = (const float*)d_in[10];  const float* Win1 = (const float*)d_in[11];
    const float* Wout1= (const float*)d_in[12];  const float* bn1 = (const float*)d_in[13];
    const float* Wg1  = (const float*)d_in[14];  const float* Wgn1 = (const float*)d_in[15];
    const float* Wge1 = (const float*)d_in[16];  const float* bg1 = (const float*)d_in[17];
    const float* We2  = (const float*)d_in[18];  const float* be2 = (const float*)d_in[19];
    const float* Wn2  = (const float*)d_in[20];  const float* Win2 = (const float*)d_in[21];
    const float* Wout2= (const float*)d_in[22];  const float* bn2 = (const float*)d_in[23];
    const float* Wg2  = (const float*)d_in[24];  const float* Wgn2 = (const float*)d_in[25];
    const float* Wge2 = (const float*)d_in[26];  const float* bg2 = (const float*)d_in[27];
    const float* Wga  = (const float*)d_in[28];  const float* bga = (const float*)d_in[29];
    const float* Wa1  = (const float*)d_in[30];  const float* ba1 = (const float*)d_in[31];
    const float* Wa2  = (const float*)d_in[32];  const float* ba2 = (const float*)d_in[33];
    const float* Wa3  = (const float*)d_in[34];  const float* ba3 = (const float*)d_in[35];
    float* out = (float*)d_out;

    char* ws = (char*)d_ws;
    size_t off = 0;
    auto alloc = [&](size_t bytes) { size_t o = off; off = alignUp256(off + bytes); return o; };

    // plain bf16 staging (row-major, stacked)
    size_t pWe1  = alloc((size_t)32 * 256 * 2);
    size_t pWn1s = alloc((size_t)576 * 256 * 2);
    size_t pWg1s = alloc((size_t)576 * 256 * 2);
    size_t pWe2  = alloc((size_t)256 * 128 * 2);
    size_t pWn2s = alloc((size_t)512 * 128 * 2);
    size_t pWg2s = alloc((size_t)512 * 128 * 2);
    size_t pWga  = alloc((size_t)128 * 128 * 2);
    size_t pWa1p = alloc((size_t)160 * 256 * 2);
    size_t pWa2  = alloc((size_t)256 * 256 * 2);
    // fragment-swizzled bf16 weights
    size_t fWe1  = alloc((size_t)32 * 256 * 2);
    size_t fWn1s = alloc((size_t)576 * 256 * 2);
    size_t fWg1s = alloc((size_t)576 * 256 * 2);
    size_t fWe2  = alloc((size_t)256 * 128 * 2);
    size_t fWn2s = alloc((size_t)512 * 128 * 2);
    size_t fWg2s = alloc((size_t)512 * 128 * 2);
    size_t fWga  = alloc((size_t)128 * 128 * 2);
    size_t fWa1p = alloc((size_t)160 * 256 * 2);
    size_t fWa2  = alloc((size_t)256 * 256 * 2);
    // accumulators / intermediates (zeroed every call)
    size_t zeroStart = off;
    size_t oInc1 = alloc((size_t)N_NODES * 256 * 4);
    size_t oOut1 = alloc((size_t)N_NODES * 256 * 4);
    size_t oInc2 = alloc((size_t)N_NODES * 128 * 4);
    size_t oOut2 = alloc((size_t)N_NODES * 128 * 4);
    size_t oN1   = alloc((size_t)N_NODES * 256 * 2);
    size_t oEg1  = alloc((size_t)N_GRAPHS * 256 * 4);
    size_t oEg2  = alloc((size_t)N_GRAPHS * 128 * 4);
    size_t oNg1  = alloc((size_t)N_GRAPHS * 256 * 4);
    size_t oNg2  = alloc((size_t)N_GRAPHS * 128 * 4);
    size_t oInd  = alloc((size_t)N_NODES * 4);
    size_t oOutd = alloc((size_t)N_NODES * 4);
    size_t oGnc  = alloc((size_t)N_GRAPHS * 4);
    size_t oGec  = alloc((size_t)N_GRAPHS * 4);
    size_t zeroEnd = off;

    unsigned short* We1p = (unsigned short*)(ws + pWe1);
    unsigned short* Wn1p = (unsigned short*)(ws + pWn1s);
    unsigned short* Wg1p = (unsigned short*)(ws + pWg1s);
    unsigned short* We2p = (unsigned short*)(ws + pWe2);
    unsigned short* Wn2p = (unsigned short*)(ws + pWn2s);
    unsigned short* Wg2p = (unsigned short*)(ws + pWg2s);
    unsigned short* Wgap = (unsigned short*)(ws + pWga);
    unsigned short* Wa1pp= (unsigned short*)(ws + pWa1p);
    unsigned short* Wa2p = (unsigned short*)(ws + pWa2);
    unsigned short* We1f = (unsigned short*)(ws + fWe1);
    unsigned short* Wn1f = (unsigned short*)(ws + fWn1s);
    unsigned short* Wg1f = (unsigned short*)(ws + fWg1s);
    unsigned short* We2f = (unsigned short*)(ws + fWe2);
    unsigned short* Wn2f = (unsigned short*)(ws + fWn2s);
    unsigned short* Wg2f = (unsigned short*)(ws + fWg2s);
    unsigned short* Wgaf = (unsigned short*)(ws + fWga);
    unsigned short* Wa1f = (unsigned short*)(ws + fWa1p);
    unsigned short* Wa2f = (unsigned short*)(ws + fWa2);
    float* inc1 = (float*)(ws + oInc1); float* out1 = (float*)(ws + oOut1);
    float* inc2 = (float*)(ws + oInc2); float* out2 = (float*)(ws + oOut2);
    unsigned short* n1 = (unsigned short*)(ws + oN1);
    float* eg1 = (float*)(ws + oEg1); float* eg2 = (float*)(ws + oEg2);
    float* ng1 = (float*)(ws + oNg1); float* ng2 = (float*)(ws + oNg2);
    int* indeg = (int*)(ws + oInd); int* outdeg = (int*)(ws + oOutd);
    int* gncnt = (int*)(ws + oGnc); int* gecnt = (int*)(ws + oGec);

    // zero accumulators (bf16 +0.0 == 0x0000, one memset covers all)
    hipMemsetAsync(ws + zeroStart, 0, zeroEnd - zeroStart, stream);

    auto cvt = [&](const float* src, unsigned short* dst, int n) {
        cvt_kernel<<<(n + 255) / 256, 256, 0, stream>>>(src, dst, n);
    };
    auto swz = [&](const unsigned short* src, unsigned short* dst, int K, int N) {
        swz_kernel<<<(K * N + 255) / 256, 256, 0, stream>>>(src, dst, K, N);
    };
    // stacked layouts: [Wn1;Win1;Wout1] etc. are contiguous row-blocks
    cvt(We1, We1p, 32 * 256);
    cvt(Wn1, Wn1p, 64 * 256); cvt(Win1, Wn1p + 64 * 256, 256 * 256);
    cvt(Wout1, Wn1p + 320 * 256, 256 * 256);
    cvt(Wg1, Wg1p, 64 * 256); cvt(Wgn1, Wg1p + 64 * 256, 256 * 256);
    cvt(Wge1, Wg1p + 320 * 256, 256 * 256);
    cvt(We2, We2p, 256 * 128);
    cvt(Wn2, Wn2p, 256 * 128); cvt(Win2, Wn2p + 256 * 128, 128 * 128);
    cvt(Wout2, Wn2p + 384 * 128, 128 * 128);
    cvt(Wg2, Wg2p, 256 * 128); cvt(Wgn2, Wg2p + 256 * 128, 128 * 128);
    cvt(Wge2, Wg2p + 384 * 128, 128 * 128);
    cvt(Wga, Wgap, 128 * 128);
    cvt(Wa1, Wa1pp, 136 * 256);
    hipMemsetAsync(Wa1pp + 136 * 256, 0, (size_t)24 * 256 * 2, stream);  // K-pad
    cvt(Wa2, Wa2p, 256 * 256);
    // swizzle into WMMA B-fragment order
    swz(We1p, We1f, 32, 256);
    swz(Wn1p, Wn1f, 576, 256);
    swz(Wg1p, Wg1f, 576, 256);
    swz(We2p, We2f, 256, 128);
    swz(Wn2p, Wn2f, 512, 128);
    swz(Wg2p, Wg2f, 512, 128);
    swz(Wgap, Wgaf, 128, 128);
    swz(Wa1pp, Wa1f, 160, 256);
    swz(Wa2p, Wa2f, 256, 256);

    // counts
    count_kernel<<<(N_EDGES + 255) / 256, 256, 0, stream>>>(
        senders, receivers, edge_graph, node_graph, indeg, outdeg, gecnt, gncnt);

    // fused edge pipeline: 31250 tiles of 16 edges
    edge_kernel<<<N_EDGES / 16, 256, 0, stream>>>(
        e, senders, receivers, edge_graph, We1f, be1, We2f, be2,
        inc1, out1, eg1, inc2, out2, eg2);

    // node blocks: 3125 tiles of 16 nodes
    node1_kernel<<<N_NODES / 16, 256, 0, stream>>>(
        x, node_graph, inc1, out1, indeg, outdeg, Wn1f, bn1, n1, ng1);
    node2_kernel<<<N_NODES / 16, 256, 0, stream>>>(
        n1, node_graph, inc2, out2, indeg, outdeg, Wn2f, bn2, ng2);

    // global pipeline: 32 tiles of 16 graphs
    global_kernel<<<N_GRAPHS / 16, 256, 0, stream>>>(
        u, aF, ng1, eg1, ng2, eg2, gncnt, gecnt,
        Wg1f, bg1, Wg2f, bg2, Wgaf, bga, Wa1f, ba1, Wa2f, ba2, Wa3, ba3, out);
}
